// SelfAttention_60962765799988
// MI455X (gfx1250) — matrix operand order repaired and checked by reading, not compiled, not run on hardware
//
#include <hip/hip_runtime.h>
#include <hip/hip_bf16.h>

// ---------------- problem constants ----------------
#define DIM_    4096
#define NH_     32
#define NKV_    8
#define HD_     128
#define BB_     2
#define SS_     2048
#define MTOT_   (BB_ * SS_)      // 4096 activation rows
#define KVD_    (NKV_ * HD_)     // 1024

typedef __attribute__((ext_vector_type(16))) __bf16 v16bf;
typedef __attribute__((ext_vector_type(8)))  __bf16 bf16x8;
typedef __attribute__((ext_vector_type(4)))  __bf16 bf16x4;
typedef __attribute__((ext_vector_type(8)))  float  v8f;
typedef __attribute__((ext_vector_type(4)))  float  f32x4;
typedef __attribute__((ext_vector_type(4)))  unsigned int v4u;
typedef __attribute__((ext_vector_type(8)))  int v8i;
typedef __attribute__((ext_vector_type(4)))  int v4i;

#if defined(__has_builtin)
#if __has_builtin(__builtin_amdgcn_tensor_load_to_lds) && __has_builtin(__builtin_amdgcn_s_wait_tensorcnt)
#define HAS_TDM 1
#endif
#endif

// =====================================================================
// TDM 2D tile load: bf16 tile (tile_d1 rows x tile_d0 elems) from a
// row-major tensor with row stride `stride_elems`, into contiguous LDS.
// Descriptor layout per CDNA5 ISA ch.8 (D# group0/group1).
// This toolchain exposes the 6-arg builtin:
//   (v4u g0, v8i g1, v4i g2, v4i g3, v8i, i32 cpol)
// =====================================================================
#ifdef HAS_TDM
__device__ __forceinline__
void tdm_load_2d(unsigned int lds_off, const __bf16* gptr,
                 unsigned int tile_d0, unsigned int tile_d1,
                 unsigned int stride_elems)
{
    unsigned long long ga = (unsigned long long)(uintptr_t)gptr;
    v4u g0;
    g0[0] = 1u;                                       // count=1, user descriptor
    g0[1] = lds_off;                                  // lds_addr (bytes)
    g0[2] = (unsigned int)ga;                         // global_addr[31:0]
    g0[3] = (unsigned int)((ga >> 32) & 0x1FFFFFFull) // global_addr[56:32]
            | (2u << 30);                             // type = 2 ("image")
    v8i g1;
    g1[0] = (int)(1u << 16);                          // data_size=1 -> 2 bytes
    g1[1] = (int)((tile_d0 & 0xFFFFu) << 16);         // tensor_dim0[15:0] @ [63:48]
    g1[2] = (int)(((tile_d0 >> 16) & 0xFFFFu)         // tensor_dim0[31:16]
            | ((tile_d1 & 0xFFFFu) << 16));           // tensor_dim1[15:0]
    g1[3] = (int)(((tile_d1 >> 16) & 0xFFFFu)         // tensor_dim1[31:16]
            | ((tile_d0 & 0xFFFFu) << 16));           // tile_dim0
    g1[4] = (int)(tile_d1 & 0xFFFFu);                 // tile_dim1 (tile_dim2=0)
    g1[5] = (int)stride_elems;                        // tensor_dim0_stride[31:0]
    g1[6] = 0;                                        // stride[47:32], dim1_stride lo
    g1[7] = 0;
    v4i z4 = {0, 0, 0, 0};
    v8i z8 = {0, 0, 0, 0, 0, 0, 0, 0};
    __builtin_amdgcn_tensor_load_to_lds(g0, g1, z4, z4, z8, 0);
}
#endif

__device__ __forceinline__ unsigned int lds_offset_of(const void* p) {
    return (unsigned int)(uintptr_t)p;     // LDS aperture: addr[31:0] = LDS byte offset
}

// =====================================================================
// One-time fp32 -> bf16 conversion (memory-bound)
// =====================================================================
__global__ void conv_bf16(const float* __restrict__ src, __bf16* __restrict__ dst,
                          size_t n4)
{
    size_t i = (size_t)blockIdx.x * blockDim.x + threadIdx.x;
    if (i >= n4) return;
    f32x4 d = *(const f32x4*)(src + i * 4);
    bf16x4 o;
#pragma unroll
    for (int c = 0; c < 4; ++c) o[c] = (__bf16)d[c];
    *(bf16x4*)(dst + i * 4) = o;
}

// =====================================================================
// One-time transpose + convert: wT[n][k] = (bf16) w[k][n].  64x64 tiles.
// =====================================================================
__global__ __launch_bounds__(256)
void transpose_conv(const float* __restrict__ w, __bf16* __restrict__ wT,
                    int K, int N)
{
    __shared__ __bf16 T[64][72];           // padded stride (144B, 16B-divisible)
    const int tid = threadIdx.x;
    const int k0 = blockIdx.y * 64;
    const int n0 = blockIdx.x * 64;
#pragma unroll
    for (int i = 0; i < 4; ++i) {
        int idx  = tid + i * 256;          // 0..1023
        int row  = idx >> 4;               // k-local 0..63
        int col4 = (idx & 15) << 2;        // n-local 0,4,..,60
        f32x4 d = *(const f32x4*)(w + (size_t)(k0 + row) * N + n0 + col4);
#pragma unroll
        for (int c = 0; c < 4; ++c) T[col4 + c][row] = (__bf16)d[c];
    }
    __syncthreads();
#pragma unroll
    for (int i = 0; i < 2; ++i) {
        int idx  = tid + i * 256;          // 0..511
        int nrow = idx >> 3;               // n-local 0..63
        int col8 = (idx & 7) << 3;         // k-local 0,8,..,56
        *(bf16x8*)(wT + (size_t)(n0 + nrow) * K + k0 + col8) =
            *(const bf16x8*)&T[nrow][col8];
    }
}

// =====================================================================
// GEMM: C(MxN) = A(MxK,bf16) * BT(NxK,bf16)^T, fp32 accumulate.
// Block 256 thr = 8 waves (4x2), block tile 256x128, wave tile 64x64,
// K-step 32.  TDM double-buffered tile loads overlapped with WMMA.
// =====================================================================
#define GBM 256
#define GBN 128
#define GBK 32

template <bool BF_OUT>
__global__ __launch_bounds__(256)
void gemm_bf16_wmma(const __bf16* __restrict__ A, const __bf16* __restrict__ BT,
                    void* __restrict__ Cout, int M, int N, int K)
{
    __shared__ __bf16 As[2][GBM][GBK];     // 2 x 16 KB
    __shared__ __bf16 Bs[2][GBN][GBK];     // 2 x  8 KB

    const int tid  = threadIdx.x;
    const int lane = tid & 31;
    const int wid  = tid >> 5;
    const int wm   = wid & 3;
    const int wn   = wid >> 2;
    const int half = lane >> 4;
    const int r    = lane & 15;
    const int m0 = blockIdx.y * GBM;
    const int n0 = blockIdx.x * GBN;

    v8f acc[4][4];
#pragma unroll
    for (int i = 0; i < 4; ++i)
#pragma unroll
        for (int j = 0; j < 4; ++j)
#pragma unroll
            for (int e = 0; e < 8; ++e) acc[i][j][e] = 0.0f;

    const int nt = K / GBK;

    // ---- tile loader: TDM DMA (preferred) or cooperative b128 copies
#ifdef HAS_TDM
#define LOAD_TILES(buf, kk0)                                                   \
    do {                                                                       \
        if (tid == 0) {                                                        \
            tdm_load_2d(lds_offset_of(&As[buf][0][0]),                         \
                        A + (size_t)m0 * K + (kk0), GBK, GBM, (unsigned)K);    \
            tdm_load_2d(lds_offset_of(&Bs[buf][0][0]),                         \
                        BT + (size_t)n0 * K + (kk0), GBK, GBN, (unsigned)K);   \
        }                                                                      \
    } while (0)
#define WAIT_TILES() do { if (tid == 0) __builtin_amdgcn_s_wait_tensorcnt(0); } while (0)
#else
#define LOAD_TILES(buf, kk0)                                                   \
    do {                                                                       \
        _Pragma("unroll")                                                      \
        for (int i = 0; i < 4; ++i) {      /* A: 256x32 = 1024 chunks of 8 */  \
            int idx = tid + i * 256; int row = idx >> 2; int c8 = (idx & 3) << 3; \
            *(bf16x8*)&As[buf][row][c8] =                                      \
                *(const bf16x8*)(A + (size_t)(m0 + row) * K + (kk0) + c8);     \
        }                                                                      \
        _Pragma("unroll")                                                      \
        for (int i = 0; i < 2; ++i) {      /* B: 128x32 = 512 chunks of 8 */   \
            int idx = tid + i * 256; int row = idx >> 2; int c8 = (idx & 3) << 3; \
            *(bf16x8*)&Bs[buf][row][c8] =                                      \
                *(const bf16x8*)(BT + (size_t)(n0 + row) * K + (kk0) + c8);    \
        }                                                                      \
    } while (0)
#define WAIT_TILES() do { } while (0)
#endif

    LOAD_TILES(0, 0);
    WAIT_TILES();
    __syncthreads();

    for (int kt = 0; kt < nt; ++kt) {
        const int cur = kt & 1, nxt = cur ^ 1;
        if (kt + 1 < nt) LOAD_TILES(nxt, (kt + 1) * GBK);

        v16bf af[4], bf[4];
#pragma unroll
        for (int i = 0; i < 4; ++i) {
            const __bf16* p = &As[cur][wm * 64 + i * 16 + r][half * 8];
            ((bf16x8*)&af[i])[0] = *(const bf16x8*)(p);
            ((bf16x8*)&af[i])[1] = *(const bf16x8*)(p + 16);
        }
#pragma unroll
        for (int j = 0; j < 4; ++j) {
            const __bf16* p = &Bs[cur][wn * 64 + j * 16 + r][half * 8];
            ((bf16x8*)&bf[j])[0] = *(const bf16x8*)(p);
            ((bf16x8*)&bf[j])[1] = *(const bf16x8*)(p + 16);
        }
#pragma unroll
        for (int i = 0; i < 4; ++i)
#pragma unroll
            for (int j = 0; j < 4; ++j)
                acc[i][j] = __builtin_amdgcn_wmma_f32_16x16x32_bf16(
                    false, af[i], false, bf[j], (short)0, acc[i][j], false, false);

        WAIT_TILES();
        __syncthreads();
    }

#pragma unroll
    for (int i = 0; i < 4; ++i)
#pragma unroll
        for (int j = 0; j < 4; ++j)
#pragma unroll
            for (int v = 0; v < 8; ++v) {
                size_t row = (size_t)(m0 + wm * 64 + i * 16 + half * 8 + v);
                size_t col = (size_t)(n0 + wn * 64 + j * 16 + r);
                if (BF_OUT) ((__bf16*)Cout)[row * N + col] = (__bf16)acc[i][j][v];
                else        ((float*) Cout)[row * N + col] = acc[i][j][v];
            }
#undef LOAD_TILES
#undef WAIT_TILES
}

// =====================================================================
// RoPE on bf16 q/k in place (fp32 math)
// =====================================================================
__global__ void rope_kernel(__bf16* __restrict__ q, __bf16* __restrict__ k,
                            const float* __restrict__ freqs)
{
    const size_t QP = (size_t)MTOT_ * NH_  * (HD_ / 2);
    const size_t KP = (size_t)MTOT_ * NKV_ * (HD_ / 2);
    size_t idx = (size_t)blockIdx.x * blockDim.x + threadIdx.x;
    if (idx >= QP + KP) return;

    __bf16* buf; int nh; size_t t;
    if (idx < QP) { buf = q; nh = NH_;  t = idx; }
    else          { buf = k; nh = NKV_; t = idx - QP; }

    const int ppr = nh * (HD_ / 2);
    size_t row = t / ppr;
    int rem = (int)(t % ppr);
    int hh  = rem / (HD_ / 2);
    int j   = rem % (HD_ / 2);
    int s   = (int)(row % SS_);

    float sn, cs;
    __sincosf(freqs[(size_t)s * (HD_ / 2) + j], &sn, &cs);

    size_t base = row * (size_t)(nh * HD_) + (size_t)hh * HD_ + 2 * j;
    float x0 = (float)buf[base], x1 = (float)buf[base + 1];
    buf[base]     = (__bf16)(x0 * cs - x1 * sn);
    buf[base + 1] = (__bf16)(x0 * sn + x1 * cs);
}

// =====================================================================
// Flash attention (non-causal, start_pos=0), GQA n_rep=4, all bf16 I/O.
// Grid (S/128, B*NH); 8 waves; wave owns 16 Q rows; KV tile = 32.
// K tile via TDM (overlapped with cooperative transposed V load).
// =====================================================================
#define FKV 32
#define ATT_SCALE 0.08838834764831845f   // 1/sqrt(128)

__global__ __launch_bounds__(256)
void flash_attn(const __bf16* __restrict__ Q, const __bf16* __restrict__ Kc,
                const __bf16* __restrict__ Vc, __bf16* __restrict__ O)
{
    __shared__ __bf16 Ks[FKV][HD_];      // 8 KB, [t][d]
    __shared__ __bf16 Vs[HD_][FKV];      // 8 KB, [d][t] (transposed)
    __shared__ __bf16 Ps[8][16][FKV];    // 8 KB, per-wave P scratch

    const int tid  = threadIdx.x;
    const int lane = tid & 31;
    const int wid  = tid >> 5;
    const int half = lane >> 4;
    const int r    = lane & 15;

    const int bh = blockIdx.y;
    const int b  = bh >> 5;
    const int h  = bh & 31;
    const int g  = h >> 2;
    const int mrow = blockIdx.x * 128 + wid * 16;

    // ---- Q fragments (bf16 direct from global, two b128 per frag)
    v16bf qf[4];
    const __bf16* qbase = Q + (size_t)(b * SS_ + mrow + r) * DIM_ + (size_t)h * HD_;
#pragma unroll
    for (int kk = 0; kk < 4; ++kk) {
        const __bf16* p = qbase + kk * 32 + half * 8;
        ((bf16x8*)&qf[kk])[0] = *(const bf16x8*)(p);
        ((bf16x8*)&qf[kk])[1] = *(const bf16x8*)(p + 16);
    }

    float mstate[8], lstate[8];
    v8f oacc[8];
#pragma unroll
    for (int v = 0; v < 8; ++v) { mstate[v] = -__builtin_inff(); lstate[v] = 0.0f; }
#pragma unroll
    for (int f = 0; f < 8; ++f)
#pragma unroll
        for (int e = 0; e < 8; ++e) oacc[f][e] = 0.0f;

    for (int t0 = 0; t0 < SS_; t0 += FKV) {
        const __bf16* kbase = Kc + (size_t)(b * SS_ + t0) * KVD_ + (size_t)g * HD_;
        const __bf16* vbase = Vc + (size_t)(b * SS_ + t0) * KVD_ + (size_t)g * HD_;
#ifdef HAS_TDM
        if (tid == 0)
            tdm_load_2d(lds_offset_of(&Ks[0][0]), kbase, HD_, FKV, KVD_);
#else
#pragma unroll
        for (int i = 0; i < 2; ++i) {
            int idx = tid + i * 256; int trow = idx >> 4; int d8 = (idx & 15) << 3;
            *(bf16x8*)&Ks[trow][d8] = *(const bf16x8*)(kbase + (size_t)trow * KVD_ + d8);
        }
#endif
        // cooperative transposed V load (overlaps the TDM DMA)
#pragma unroll
        for (int i = 0; i < 2; ++i) {
            int idx = tid + i * 256; int trow = idx >> 4; int d8 = (idx & 15) << 3;
            bf16x8 vd = *(const bf16x8*)(vbase + (size_t)trow * KVD_ + d8);
#pragma unroll
            for (int c = 0; c < 8; ++c) Vs[d8 + c][trow] = vd[c];
        }
#ifdef HAS_TDM
        if (tid == 0) __builtin_amdgcn_s_wait_tensorcnt(0);
#endif
        __syncthreads();

        // ---- scores: 16x32 in two accumulators
        v8f sacc[2];
#pragma unroll
        for (int jn = 0; jn < 2; ++jn)
#pragma unroll
            for (int e = 0; e < 8; ++e) sacc[jn][e] = 0.0f;
#pragma unroll
        for (int kk = 0; kk < 4; ++kk)
#pragma unroll
            for (int jn = 0; jn < 2; ++jn) {
                const __bf16* p = &Ks[jn * 16 + r][kk * 32 + half * 8];
                v16bf bfk;
                ((bf16x8*)&bfk)[0] = *(const bf16x8*)(p);
                ((bf16x8*)&bfk)[1] = *(const bf16x8*)(p + 16);
                sacc[jn] = __builtin_amdgcn_wmma_f32_16x16x32_bf16(
                    false, qf[kk], false, bfk, (short)0, sacc[jn], false, false);
            }

        // ---- online softmax (fp32; v_exp_f32 + lane shuffles)
#pragma unroll
        for (int v = 0; v < 8; ++v) {
            float a  = sacc[0][v] * ATT_SCALE;
            float bb = sacc[1][v] * ATT_SCALE;
            float mv = fmaxf(a, bb);
#pragma unroll
            for (int off = 1; off < 16; off <<= 1)
                mv = fmaxf(mv, __shfl_xor(mv, off, 32));
            float newm = fmaxf(mstate[v], mv);
            float fac  = __expf(mstate[v] - newm);
            float p0 = __expf(a - newm);
            float p1 = __expf(bb - newm);
            float sum = p0 + p1;
#pragma unroll
            for (int off = 1; off < 16; off <<= 1)
                sum += __shfl_xor(sum, off, 32);
            lstate[v] = lstate[v] * fac + sum;
            mstate[v] = newm;
#pragma unroll
            for (int f = 0; f < 8; ++f) oacc[f][v] *= fac;
            Ps[wid][half * 8 + v][r]      = (__bf16)p0;
            Ps[wid][half * 8 + v][16 + r] = (__bf16)p1;
        }

        // ---- O += P * V
        const __bf16* pp = &Ps[wid][r][half * 8];
        v16bf pf;
        ((bf16x8*)&pf)[0] = *(const bf16x8*)(pp);
        ((bf16x8*)&pf)[1] = *(const bf16x8*)(pp + 16);
#pragma unroll
        for (int f = 0; f < 8; ++f) {
            const __bf16* vp = &Vs[f * 16 + r][half * 8];
            v16bf vf;
            ((bf16x8*)&vf)[0] = *(const bf16x8*)(vp);
            ((bf16x8*)&vf)[1] = *(const bf16x8*)(vp + 16);
            oacc[f] = __builtin_amdgcn_wmma_f32_16x16x32_bf16(
                false, pf, false, vf, (short)0, oacc[f], false, false);
        }
        __syncthreads();
    }

    float inv[8];
#pragma unroll
    for (int v = 0; v < 8; ++v) inv[v] = 1.0f / lstate[v];
#pragma unroll
    for (int f = 0; f < 8; ++f)
#pragma unroll
        for (int v = 0; v < 8; ++v) {
            size_t row = (size_t)(b * SS_ + mrow + half * 8 + v);
            O[row * DIM_ + (size_t)h * HD_ + f * 16 + r] = (__bf16)(oacc[f][v] * inv[v]);
        }
}

// =====================================================================
// launch
// =====================================================================
extern "C" void kernel_launch(void* const* d_in, const int* in_sizes, int n_in,
                              void* d_out, int out_size, void* d_ws, size_t ws_size,
                              hipStream_t stream) {
    const float* x     = (const float*)d_in[0];
    const float* freqs = (const float*)d_in[1];
    const float* wq    = (const float*)d_in[2];
    const float* wk    = (const float*)d_in[3];
    const float* wv    = (const float*)d_in[4];
    const float* wo    = (const float*)d_in[5];
    // d_in[6] = start_pos (0 in reference setup; cache empty -> span [0,S))
    float* out = (float*)d_out;

    // ---- workspace (all bf16): 192 MB total
    __bf16* xb   = (__bf16*)d_ws;
    __bf16* wqT  = xb   + (size_t)MTOT_ * DIM_;
    __bf16* wkT  = wqT  + (size_t)DIM_ * DIM_;
    __bf16* wvT  = wkT  + (size_t)KVD_ * DIM_;
    __bf16* woT  = wvT  + (size_t)KVD_ * DIM_;
    __bf16* qb   = woT  + (size_t)DIM_ * DIM_;
    __bf16* kb   = qb   + (size_t)MTOT_ * DIM_;
    __bf16* vb   = kb   + (size_t)MTOT_ * KVD_;
    __bf16* attn = vb   + (size_t)MTOT_ * KVD_;

    // ---- one-time conversions / transposes (memory-bound)
    {
        size_t n4 = (size_t)MTOT_ * DIM_ / 4;
        conv_bf16<<<(int)((n4 + 255) / 256), 256, 0, stream>>>(x, xb, n4);
    }
    transpose_conv<<<dim3(DIM_ / 64, DIM_ / 64), 256, 0, stream>>>(wq, wqT, DIM_, DIM_);
    transpose_conv<<<dim3(KVD_ / 64, DIM_ / 64), 256, 0, stream>>>(wk, wkT, DIM_, KVD_);
    transpose_conv<<<dim3(KVD_ / 64, DIM_ / 64), 256, 0, stream>>>(wv, wvT, DIM_, KVD_);
    transpose_conv<<<dim3(DIM_ / 64, DIM_ / 64), 256, 0, stream>>>(wo, woT, DIM_, DIM_);

    // ---- QKV projections (bf16 WMMA + TDM double buffering)
    gemm_bf16_wmma<true><<<dim3(DIM_ / GBN, MTOT_ / GBM), 256, 0, stream>>>(
        xb, wqT, qb, MTOT_, DIM_, DIM_);
    gemm_bf16_wmma<true><<<dim3(KVD_ / GBN, MTOT_ / GBM), 256, 0, stream>>>(
        xb, wkT, kb, MTOT_, KVD_, DIM_);
    gemm_bf16_wmma<true><<<dim3(KVD_ / GBN, MTOT_ / GBM), 256, 0, stream>>>(
        xb, wvT, vb, MTOT_, KVD_, DIM_);

    // ---- RoPE in place
    {
        size_t total = (size_t)MTOT_ * NH_ * (HD_ / 2) + (size_t)MTOT_ * NKV_ * (HD_ / 2);
        rope_kernel<<<(int)((total + 255) / 256), 256, 0, stream>>>(qb, kb, freqs);
    }

    // ---- flash attention
    flash_attn<<<dim3(SS_ / 128, BB_ * NH_), 256, 0, stream>>>(qb, kb, vb, attn);

    // ---- output projection (fp32 out)
    gemm_bf16_wmma<false><<<dim3(DIM_ / GBN, MTOT_ / GBM), 256, 0, stream>>>(
        attn, woT, out, MTOT_, DIM_, DIM_);
}
